// GraphAttn_49039936586130
// MI455X (gfx1250) — matrix operand order repaired and checked by reading, not compiled, not run on hardware
//
#include <hip/hip_runtime.h>

typedef __attribute__((ext_vector_type(16))) __bf16 v16bf;
typedef __attribute__((ext_vector_type(8)))  __bf16 v8bf;
typedef __attribute__((ext_vector_type(8)))  float  v8f;

#define N_NODES 8192
#define F_IN    256
#define F_OUT   128
#define ALPHA   0.2f

// LDS row strides (elements), padded to avoid 64-bank conflicts on b128 loads
#define IN_STR  264   // bf16, 16x256 input tile
#define H_STR   136   // bf16, 16x128 activation tiles
#define F_STR   132   // f32,  16x128 final-h tile

// workspace layout (bytes)
#define WPACK_ELEMS (8*8*512 + 3*(4*8*512))     // 81920 bf16
#define WPACK_BYTES (WPACK_ELEMS * 2)           // 163840
#define HPACK_ELEMS (256 * 8 * 512)             // 1048576 bf16 (h in B-packed layout)
#define HPACK_BYTES (HPACK_ELEMS * 2)           // 2097152
#define W2_OFF 32768
#define W3_OFF 49152
#define W4_OFF 65536

static __device__ __forceinline__ v16bf cat16(v8bf lo, v8bf hi) {
    return __builtin_shufflevector(lo, hi, 0,1,2,3,4,5,6,7,8,9,10,11,12,13,14,15);
}

// ---------------------------------------------------------------------------
// Repack W (K x 128 row-major f32) into WMMA B-operand order, bf16.
// B tile (kt,nt): lane holds col n = nt*16+(lane&15), K = kt*32+(lane>>4)*8+(i&7)+(i>>3)*16
// packed offset: (kt*8+nt)*512 + lane*16 + i
// ---------------------------------------------------------------------------
__global__ __launch_bounds__(256) void pack_weights(const float* __restrict__ W1,
                                                    const float* __restrict__ W2,
                                                    const float* __restrict__ W3,
                                                    const float* __restrict__ W4,
                                                    __bf16* __restrict__ wpack) {
    int p = blockIdx.x * blockDim.x + threadIdx.x;
    if (p >= WPACK_ELEMS) return;
    const float* W; int base;
    if (p < W2_OFF)      { W = W1; base = 0;      }
    else if (p < W3_OFF) { W = W2; base = W2_OFF; }
    else if (p < W4_OFF) { W = W3; base = W3_OFF; }
    else                 { W = W4; base = W4_OFF; }
    int q    = p - base;
    int tile = q >> 9;            // /512
    int lane = (q >> 4) & 31;
    int i    = q & 15;
    int kt = tile >> 3, nt = tile & 7;
    int k  = kt * 32 + ((lane >> 4) << 3) + (i & 7) + ((i >> 3) << 4);
    int n  = (nt << 4) + (lane & 15);
    wpack[p] = (__bf16)W[k * F_OUT + n];
}

// ---------------------------------------------------------------------------
// One dense layer on a 16-row tile: out(16x128) = act(in(16xKD) @ Wpack + b)
// in/out are LDS row-major bf16. Optionally mirror f32 result to outf.
// ---------------------------------------------------------------------------
static __device__ __forceinline__ void layer_wmma(const __bf16* in, int in_str, int Kd,
                                                  const __bf16* __restrict__ wseg,
                                                  const float* __restrict__ bias,
                                                  __bf16* outb, float* outf,
                                                  bool relu, int lane) {
    const int row = lane & 15, g = lane >> 4, off = g * 8, col = lane & 15;
    const int nkt = Kd >> 5;
    for (int nt = 0; nt < 8; ++nt) {
        float bv = bias[nt * 16 + col];
        v8f acc = {bv, bv, bv, bv, bv, bv, bv, bv};
        for (int kt = 0; kt < nkt; ++kt) {
            const __bf16* ap = in + row * in_str + kt * 32 + off;
            v16bf A = cat16(*(const v8bf*)ap, *(const v8bf*)(ap + 16));
            const v8bf* bp = (const v8bf*)(wseg + ((size_t)(kt * 8 + nt) << 9) + lane * 16);
            v16bf B = cat16(bp[0], bp[1]);
            acc = __builtin_amdgcn_wmma_f32_16x16x32_bf16(false, A, false, B,
                                                          (short)0, acc, false, false);
        }
#pragma unroll
        for (int v = 0; v < 8; ++v) {
            float x = acc[v];
            if (relu) x = x > 0.f ? x : 0.f;
            int m = v + 8 * g;
            outb[m * H_STR + nt * 16 + col] = (__bf16)x;
            if (outf) outf[m * F_STR + nt * 16 + col] = x;
        }
    }
}

// ---------------------------------------------------------------------------
// MLP: h = (relu∘fc)x3 then fc; emit h in B-packed bf16 + s_src/s_dst f32.
// One wave per block, 16 rows per wave.
// ---------------------------------------------------------------------------
__global__ __launch_bounds__(32) void mlp_kernel(const float* __restrict__ nodes,
                                                 const float* __restrict__ b1,
                                                 const float* __restrict__ b2,
                                                 const float* __restrict__ b3,
                                                 const float* __restrict__ b4,
                                                 const float* __restrict__ a_w,
                                                 const __bf16* __restrict__ wpack,
                                                 __bf16* __restrict__ hpack,
                                                 float* __restrict__ s_src,
                                                 float* __restrict__ s_dst) {
    __shared__ __align__(16) __bf16 sIn[16 * IN_STR];
    __shared__ __align__(16) __bf16 sPing[16 * H_STR];
    __shared__ __align__(16) __bf16 sPong[16 * H_STR];
    __shared__ __align__(16) float  sH[16 * F_STR];

    const int lane = threadIdx.x;
    const int r0   = blockIdx.x * 16;

    // stage input rows as bf16 into LDS (float4 granularity)
    for (int t4 = lane; t4 < 16 * (F_IN / 4); t4 += 32) {
        int r = t4 >> 6, c4 = t4 & 63;
        float4 x = ((const float4*)(nodes + (size_t)(r0 + r) * F_IN))[c4];
        __bf16* dst = sIn + r * IN_STR + c4 * 4;
        dst[0] = (__bf16)x.x; dst[1] = (__bf16)x.y;
        dst[2] = (__bf16)x.z; dst[3] = (__bf16)x.w;
    }
    __syncthreads();

    layer_wmma(sIn,   IN_STR, 256, wpack,          b1, sPing, nullptr, true,  lane);
    __syncthreads();
    layer_wmma(sPing, H_STR,  128, wpack + W2_OFF, b2, sPong, nullptr, true,  lane);
    __syncthreads();
    layer_wmma(sPong, H_STR,  128, wpack + W3_OFF, b3, sPing, nullptr, true,  lane);
    __syncthreads();
    layer_wmma(sPing, H_STR,  128, wpack + W4_OFF, b4, sPong, sH,      false, lane);
    __syncthreads();

    // attention scalars: lanes 0-15 -> s_src (a_w[0:128]); lanes 16-31 -> s_dst
    {
        int r = lane & 15;
        const float* av = a_w + ((lane < 16) ? 0 : F_OUT);
        float s = 0.f;
        for (int n = 0; n < F_OUT; ++n) s += sH[r * F_STR + n] * av[n];
        if (lane < 16) s_src[r0 + r] = s;
        else           s_dst[r0 + r] = s;
    }

    // emit h in B-packed layout: hpack[(jt*8+nt)*512 + plane*16 + i]
    //   = h[jt*32 + (plane>>4)*8 + (i&7) + (i>>3)*16 , nt*16 + (plane&15)]
    {
        int jt = r0 >> 5, half = (r0 >> 4) & 1;
        for (int q = 0; q < 4; ++q) {
            int c = lane * 4 + q;            // 0..127 chunk id
            int nt = c >> 4, sN = c & 15;
            int plane = sN;
            __bf16* dst = hpack + (((size_t)(jt * 8 + nt)) << 9) + plane * 16;
            v8bf o0, o1;
#pragma unroll
            for (int i = 0; i < 8; ++i) o0[i] = sPong[i * H_STR + nt * 16 + sN];
#pragma unroll
            for (int i = 0; i < 8; ++i) o1[i] = sPong[(i + 8) * H_STR + nt * 16 + sN];
            *(v8bf*)(dst + half * 8)       = o0;
            *(v8bf*)(dst + 256 + half * 8) = o1;
        }
    }
}

// ---------------------------------------------------------------------------
// Fused masked-softmax aggregation: single pass over adj.
// 512 blocks x 128 threads; 4 waves split the j-range of one 16-row tile,
// partial (acc, L) merged via LDS float atomics, then out = leaky(acc/L).
// ---------------------------------------------------------------------------
__global__ __launch_bounds__(128) void agg_kernel(const float* __restrict__ adj,
                                                  const __bf16* __restrict__ hpack,
                                                  const float* __restrict__ s_src,
                                                  const float* __restrict__ s_dst,
                                                  const float* __restrict__ a_bp,
                                                  float* __restrict__ out) {
    __shared__ __align__(16) float sAcc[16 * 128];
    __shared__ float sL[16];

    const int t    = threadIdx.x;
    const int lane = t & 31;
    const int wv   = t >> 5;                 // j-split index 0..3
    const int r0   = blockIdx.x * 16;
    const int row  = lane & 15;              // A row / C column index
    const int g    = lane >> 4;
    const int off  = g * 8;

    for (int i = t; i < 16 * 128; i += 128) sAcc[i] = 0.f;
    if (t < 16) sL[t] = 0.f;
    __syncthreads();

    const float s_me = s_src[r0 + row];
    const float ab   = a_bp[0];
    const float* arow = adj + (size_t)(r0 + row) * N_NODES;

    v8f acc[8];
    v8f z = {0, 0, 0, 0, 0, 0, 0, 0};
#pragma unroll
    for (int nt = 0; nt < 8; ++nt) acc[nt] = z;
    float Lp = 0.f;

    const int kt0 = wv * 64, kt1 = kt0 + 64;
    for (int kt = kt0; kt < kt1; ++kt) {
        const int jb = kt << 5;
        const float4* ap = (const float4*)(arow + jb + off);
        const float4* dp = (const float4*)(s_dst + jb + off);
        float4 m0 = ap[0], m1 = ap[1], m2 = ap[4], m3 = ap[5];
        float4 d0 = dp[0], d1 = dp[1], d2 = dp[4], d3 = dp[5];

        float p[16];
        auto prob = [&](float mv, float dv) -> float {
            float e = s_me + dv + ab;
            e = (e >= 0.f) ? e : ALPHA * e;          // leaky
            float pv = (mv >= 0.5f) ? __expf(e) : 0.f; // mask + exp (logits tiny: safe)
            Lp += pv;
            return pv;
        };
        p[0]=prob(m0.x,d0.x);  p[1]=prob(m0.y,d0.y);  p[2]=prob(m0.z,d0.z);  p[3]=prob(m0.w,d0.w);
        p[4]=prob(m1.x,d1.x);  p[5]=prob(m1.y,d1.y);  p[6]=prob(m1.z,d1.z);  p[7]=prob(m1.w,d1.w);
        p[8]=prob(m2.x,d2.x);  p[9]=prob(m2.y,d2.y);  p[10]=prob(m2.z,d2.z); p[11]=prob(m2.w,d2.w);
        p[12]=prob(m3.x,d3.x); p[13]=prob(m3.y,d3.y); p[14]=prob(m3.z,d3.z); p[15]=prob(m3.w,d3.w);

        v16bf A;
#pragma unroll
        for (int i = 0; i < 16; ++i) A[i] = (__bf16)p[i];

#pragma unroll
        for (int nt = 0; nt < 8; ++nt) {
            const v8bf* bp = (const v8bf*)(hpack + (((size_t)(kt * 8 + nt)) << 9) + lane * 16);
            v16bf B = cat16(bp[0], bp[1]);
            acc[nt] = __builtin_amdgcn_wmma_f32_16x16x32_bf16(false, A, false, B,
                                                              (short)0, acc[nt], false, false);
        }
    }

    // merge partials across the 4 j-split waves
    atomicAdd(&sL[row], Lp);
#pragma unroll
    for (int nt = 0; nt < 8; ++nt)
#pragma unroll
        for (int v = 0; v < 8; ++v) {
            int m = v + 8 * g;
            atomicAdd(&sAcc[m * 128 + nt * 16 + row], acc[nt][v]);
        }
    __syncthreads();

    // normalize + leaky + store
    for (int i = t; i < 16 * 128; i += 128) {
        int m = i >> 7;
        float x = sAcc[i] / sL[m];
        x = (x >= 0.f) ? x : ALPHA * x;
        out[(size_t)(r0 + m) * F_OUT + (i & 127)] = x;
    }
}

// ---------------------------------------------------------------------------
extern "C" void kernel_launch(void* const* d_in, const int* in_sizes, int n_in,
                              void* d_out, int out_size, void* d_ws, size_t ws_size,
                              hipStream_t stream) {
    (void)in_sizes; (void)n_in; (void)out_size;
    const float* nodes = (const float*)d_in[0];
    const float* adj   = (const float*)d_in[1];
    const float* W1    = (const float*)d_in[2];
    const float* b1    = (const float*)d_in[3];
    const float* W2    = (const float*)d_in[4];
    const float* b2    = (const float*)d_in[5];
    const float* W3    = (const float*)d_in[6];
    const float* b3    = (const float*)d_in[7];
    const float* W4    = (const float*)d_in[8];
    const float* b4    = (const float*)d_in[9];
    const float* a_w   = (const float*)d_in[10];
    const float* a_b   = (const float*)d_in[11];
    float* out = (float*)d_out;

    char* ws = (char*)d_ws;
    __bf16* wpack = (__bf16*)(ws);
    __bf16* hpack = (__bf16*)(ws + WPACK_BYTES);
    float*  s_src = (float*)(ws + WPACK_BYTES + HPACK_BYTES);
    float*  s_dst = s_src + N_NODES;
    if (ws_size < (size_t)WPACK_BYTES + HPACK_BYTES + 2u * N_NODES * 4u) return;

    hipLaunchKernelGGL(pack_weights, dim3((WPACK_ELEMS + 255) / 256), dim3(256), 0, stream,
                       W1, W2, W3, W4, wpack);
    hipLaunchKernelGGL(mlp_kernel, dim3(N_NODES / 16), dim3(32), 0, stream,
                       nodes, b1, b2, b3, b4, a_w, wpack, hpack, s_src, s_dst);
    hipLaunchKernelGGL(agg_kernel, dim3(N_NODES / 16), dim3(128), 0, stream,
                       adj, hpack, s_src, s_dst, a_b, out);
}
